// EncoderLayer_6536940224870
// MI455X (gfx1250) — matrix operand order repaired and checked by reading, not compiled, not run on hardware
//
#include <hip/hip_runtime.h>

typedef __attribute__((ext_vector_type(16))) _Float16 v16h;
typedef __attribute__((ext_vector_type(8)))  _Float16 v8h;
typedef __attribute__((ext_vector_type(8)))  float    v8f;
typedef __attribute__((vector_size(16)))     int      v4i;   // for async LDS-DMA builtin

#define B_ 4
#define L_ 201
#define C_ 55
#define D_ 512
#define F_ 2048
#define N_ 200          // L-1 temporal tokens
#define KT_ 20          // cluster_temporal
#define KC_ 10          // cluster_channel
#define NP_ 208         // padded temporal rows (mult 16)
#define CP_ 64          // padded channel rows
#define TPAD_ 224       // padded K-dim for P@V (mult 32, >= 201)
#define ROWS_CAP 46080  // slack rows for flat f16 activation buffers (covers tile overreads)
#define MFLAT 44224     // B*L*C padded to mult 16 (and 32)
#define MT1 45760       // B*C*208
#define MX2 51456       // B*L*64
#define INV_SQRT_D 0.044194173824159216f

#define EP_F16  1
#define EP_BIAS 2
#define EP_GELU 4

// ---- CDNA5 async LDS-DMA availability (device pass only; host pass falls back)
#ifdef __HIP_DEVICE_COMPILE__
#if defined(__has_builtin)
#if __has_builtin(__builtin_amdgcn_global_load_async_to_lds_b128) && \
    __has_builtin(__builtin_amdgcn_s_wait_asynccnt)
#define HAVE_ASYNC_LDS 1
#endif
#endif
#endif

// ---------------------------------------------------------------- utilities
__device__ __forceinline__ float hash01(unsigned s) {
  s ^= s >> 16; s *= 0x7feb352du; s ^= s >> 15; s *= 0x846ca68bu; s ^= s >> 16;
  return (float)(s >> 8) * (1.0f / 16777216.0f);
}

__global__ void fill_f32k(float* p, long long n, float v) {
  long long i = blockIdx.x * 256ll + threadIdx.x;
  if (i < n) p[i] = v;
}
__global__ void fill_f16k(_Float16* p, long long n) {
  long long i = blockIdx.x * 256ll + threadIdx.x;
  if (i < n) p[i] = (_Float16)0.f;
}

// W [K,N] f32 -> Wt [N,K] f16, 64x64 tiles staged through LDS.
// Uses CDNA5 async global->LDS DMA (ASYNCcnt) when the toolchain exposes it.
__global__ void __launch_bounds__(256)
wt_transpose_lds(const float* __restrict__ W, _Float16* __restrict__ Wt, int K, int N) {
  __shared__ float tile[64][68];
  int kt = blockIdx.x % (K >> 6);
  int nt = blockIdx.x / (K >> 6);
  int tid = threadIdx.x;
#ifdef HAVE_ASYNC_LDS
#pragma unroll
  for (int q = 0; q < 4; q++) {
    int unit = tid + q * 256;            // 0..1023 : 16B chunks of the 64x64 tile
    int kk = unit >> 4;
    int jg = unit & 15;
    const float* src = W + (long long)(kt * 64 + kk) * N + nt * 64 + jg * 4;
    void* dst = (void*)&tile[kk][jg * 4];
    __builtin_amdgcn_global_load_async_to_lds_b128((v4i*)src, (v4i*)dst, 0, 0);
  }
  __builtin_amdgcn_s_wait_asynccnt(0);
#else
#pragma unroll
  for (int q = 0; q < 4; q++) {
    int unit = tid + q * 256;
    int kk = unit >> 4;
    int jg = unit & 15;
    const float* src = W + (long long)(kt * 64 + kk) * N + nt * 64 + jg * 4;
#pragma unroll
    for (int e = 0; e < 4; e++) tile[kk][jg * 4 + e] = src[e];
  }
#endif
  __syncthreads();
#pragma unroll
  for (int q = 0; q < 2; q++) {
    int unit = tid + q * 256;            // 0..511 : 8-half output chunks
    int n = unit & 63;
    int kg = unit >> 6;                  // 0..7
    v8h vals;
#pragma unroll
    for (int e = 0; e < 8; e++) vals[e] = (_Float16)tile[kg * 8 + e][n];
    *(v8h*)(Wt + (long long)(nt * 64 + n) * K + kt * 64 + kg * 8) = vals;
  }
}

// ------------------------------------------------- WMMA fragment load helpers
// A 16x32 f16 frag: lanes 0-15 -> K {0..7,16..23}; lanes 16-31 -> K {8..15,24..31}
__device__ __forceinline__ v16h ldA(const _Float16* p, int hi) {
  v8h x0 = *(const v8h*)(p + hi * 8);
  v8h x1 = *(const v8h*)(p + 16 + hi * 8);
  v16h v;
#pragma unroll
  for (int q = 0; q < 8; q++) { v[q] = x0[q]; v[q + 8] = x1[q]; }
  return v;
}
// B 32x16 f16 frag: lanes 0-15 -> K {0..7,16..23}; lanes 16-31 -> K {8..15,24..31}
__device__ __forceinline__ v16h ldB(const _Float16* p, int hi) {
  v8h x0 = *(const v8h*)(p + hi * 8);
  v8h x1 = *(const v8h*)(p + 16 + hi * 8);
  v16h v;
#pragma unroll
  for (int q = 0; q < 8; q++) { v[q] = x0[q]; v[q + 8] = x1[q]; }
  return v;
}
__device__ __forceinline__ void epi_store(void* Cout, const float* bias, v8f acc,
                                          int row0, int col, int ldc, int flags,
                                          float alpha, int hi) {
  float bi = (flags & EP_BIAS) ? bias[col] : 0.f;
#pragma unroll
  for (int i = 0; i < 8; i++) {
    int row = row0 + i + hi * 8;   // C/D layout: vgpr i -> M=i (lanes<16) / M=8+i
    float v = acc[i] * alpha + bi;
    if (flags & EP_GELU) v = 0.5f * v * (1.f + erff(v * 0.70710678f));
    if (flags & EP_F16) ((_Float16*)Cout)[(long long)row * ldc + col] = (_Float16)v;
    else                ((float*)Cout)[(long long)row * ldc + col] = v;
  }
}

// ------------------------- generic batched WMMA GEMM (16x16/wave, NT, f16->f32)
__global__ void __launch_bounds__(256)
gemm_nt(const _Float16* __restrict__ A, const _Float16* __restrict__ Bt,
        void* __restrict__ Cout, const float* __restrict__ bias,
        int Mt, int Nt, int K, int lda, int ldb, int ldc,
        long long aOuter, long long aInner, int aInnerN,
        long long bOuter, long long bInner, int bInnerN,
        long long cStride, int batches, int flags, float alpha) {
  int wave = blockIdx.x * (blockDim.x >> 5) + (threadIdx.x >> 5);
  int total = batches * Mt * Nt;
  if (wave >= total) return;
  int bb = wave / (Mt * Nt);
  int rem = wave % (Mt * Nt);
  int mt = rem / Nt, nt = rem % Nt;
  int lane = threadIdx.x & 31;
  int r = lane & 15, hi = lane >> 4;
  const _Float16* Ap = A + (long long)(bb / aInnerN) * aOuter
                         + (long long)(bb % aInnerN) * aInner
                         + (long long)(mt * 16 + r) * lda;
  const _Float16* Bp = Bt + (long long)(bb / bInnerN) * bOuter
                          + (long long)(bb % bInnerN) * bInner
                          + (long long)(nt * 16 + r) * ldb;
  v8f acc = {};
  for (int k = 0; k < K; k += 32) {
    v16h av = ldA(Ap + k, hi);
    v16h bv = ldB(Bp + k, hi);
    acc = __builtin_amdgcn_wmma_f32_16x16x32_f16(false, av, false, bv, (short)0, acc, false, false);
    __builtin_prefetch(Ap + k + 128, 0, 0);
    __builtin_prefetch(Bp + k + 128, 0, 0);
  }
  long long cbase = (long long)bb * cStride;
  void* Cb = (flags & EP_F16) ? (void*)((_Float16*)Cout + cbase) : (void*)((float*)Cout + cbase);
  epi_store(Cb, bias, acc, mt * 16, nt * 16 + r, ldc, flags, alpha, hi);
}

// ---------------- register-blocked flat WMMA GEMM (32x32/wave, 2x2 tiles)
__global__ void __launch_bounds__(256)
gemm_nt2(const _Float16* __restrict__ A, const _Float16* __restrict__ Bt,
         void* __restrict__ Cout, const float* __restrict__ bias,
         int Mt2, int Nt2, int K, int lda, int ldb, int ldc,
         int flags, float alpha) {
  int wave = blockIdx.x * (blockDim.x >> 5) + (threadIdx.x >> 5);
  if (wave >= Mt2 * Nt2) return;
  int mt = wave / Nt2, nt = wave % Nt2;
  int lane = threadIdx.x & 31;
  int r = lane & 15, hi = lane >> 4;
  const _Float16* A0 = A + (long long)(mt * 32 + r) * lda;
  const _Float16* A1 = A0 + (long long)16 * lda;
  const _Float16* B0 = Bt + (long long)(nt * 32 + r) * ldb;
  const _Float16* B1 = B0 + (long long)16 * ldb;
  v8f acc00 = {}, acc01 = {}, acc10 = {}, acc11 = {};
  for (int k = 0; k < K; k += 32) {
    v16h a0 = ldA(A0 + k, hi);
    v16h a1 = ldA(A1 + k, hi);
    v16h b0 = ldB(B0 + k, hi);
    v16h b1 = ldB(B1 + k, hi);
    acc00 = __builtin_amdgcn_wmma_f32_16x16x32_f16(false, a0, false, b0, (short)0, acc00, false, false);
    acc01 = __builtin_amdgcn_wmma_f32_16x16x32_f16(false, a0, false, b1, (short)0, acc01, false, false);
    acc10 = __builtin_amdgcn_wmma_f32_16x16x32_f16(false, a1, false, b0, (short)0, acc10, false, false);
    acc11 = __builtin_amdgcn_wmma_f32_16x16x32_f16(false, a1, false, b1, (short)0, acc11, false, false);
    __builtin_prefetch(A0 + k + 128, 0, 0);
    __builtin_prefetch(B0 + k + 128, 0, 0);
  }
  epi_store(Cout, bias, acc00, mt * 32,      nt * 32 + r,      ldc, flags, alpha, hi);
  epi_store(Cout, bias, acc01, mt * 32,      nt * 32 + 16 + r, ldc, flags, alpha, hi);
  epi_store(Cout, bias, acc10, mt * 32 + 16, nt * 32 + r,      ldc, flags, alpha, hi);
  epi_store(Cout, bias, acc11, mt * 32 + 16, nt * 32 + 16 + r, ldc, flags, alpha, hi);
}

// --------------------------------------------------------- clustering kernels
__global__ void build_xch_t(const float* __restrict__ x, _Float16* __restrict__ xch) {
  long long i = blockIdx.x * 256ll + threadIdx.x;
  const long long total = (long long)C_ * B_ * NP_ * D_;
  if (i >= total) return;
  int d = (int)(i & 511); long long r = i >> 9;
  int n = (int)(r % NP_); int bb = (int)(r / NP_);
  int b = bb & 3, c = bb >> 2;           // bb = c*B + b
  float v = 0.f;
  if (n < N_) v = x[(((long long)b * L_ + 1 + n) * C_ + c) * D_ + d];
  xch[i] = (_Float16)v;
}
__global__ void build_xc0(const float* __restrict__ x, _Float16* __restrict__ o) {
  int i = blockIdx.x * 256 + threadIdx.x;
  const int total = B_ * CP_ * D_;
  if (i >= total) return;
  int d = i & 511; int r = i >> 9; int c = r % CP_; int b = r / CP_;
  float v = (c < C_) ? x[(((long long)b * L_) * C_ + c) * D_ + d] : 0.f;
  o[i] = (_Float16)v;
}
__global__ void dist_from_gram(const float* __restrict__ G, float* __restrict__ dist,
                               int batches, int Nv, int ld) {
  long long i = blockIdx.x * 256ll + threadIdx.x;
  long long total = (long long)batches * ld * ld;
  if (i >= total) return;
  int j = (int)(i % ld); long long r = i / ld;
  int ii = (int)(r % ld); int bb = (int)(r / ld);
  const float* g = G + (long long)bb * ld * ld;
  float v;
  if (ii < Nv && j < Nv) {
    float d2 = g[(long long)ii * ld + ii] + g[(long long)j * ld + j] - 2.f * g[(long long)ii * ld + j];
    v = sqrtf(fmaxf(d2, 0.f)) * INV_SQRT_D;
  } else v = 1e30f;
  dist[i] = v;
}
__global__ void knn_density(const float* __restrict__ dist, float* __restrict__ dens,
                            int bbCount, int Nv, int ld, int kk, unsigned seed) {
  int t = blockIdx.x * 256 + threadIdx.x;
  if (t >= bbCount * Nv) return;
  int bb = t / Nv, i = t % Nv;
  const float* row = dist + (long long)bb * ld * ld + (long long)i * ld;
  float best[20];
#pragma unroll
  for (int q = 0; q < 20; q++) best[q] = 3e38f;
  for (int j = 0; j < Nv; j++) {
    float d = row[j];
    if (d < best[kk - 1]) {
      int p = kk - 1;
      while (p > 0 && best[p - 1] > d) { best[p] = best[p - 1]; p--; }
      best[p] = d;
    }
  }
  float s = 0.f;
  for (int q = 0; q < kk; q++) s += best[q] * best[q];
  dens[(long long)bb * ld + i] = __expf(-s / kk) + hash01((unsigned)t * 2654435761u ^ seed) * 1e-6f;
}
__global__ void reduce_maxk(const float* __restrict__ dist, float* __restrict__ dmax,
                            int Nv, int ld) {
  long long base = (long long)blockIdx.x * ld * ld;
  float m = 0.f;
  for (int idx = threadIdx.x; idx < Nv * Nv; idx += blockDim.x) {
    int i = idx / Nv, j = idx % Nv;
    m = fmaxf(m, dist[base + (long long)i * ld + j]);
  }
  __shared__ float sm[256];
  sm[threadIdx.x] = m; __syncthreads();
  for (int s = 128; s > 0; s >>= 1) {
    if ((int)threadIdx.x < s) sm[threadIdx.x] = fmaxf(sm[threadIdx.x], sm[threadIdx.x + s]);
    __syncthreads();
  }
  if (threadIdx.x == 0) dmax[blockIdx.x] = sm[0];
}
__global__ void dparent_score(const float* __restrict__ dist, const float* __restrict__ dens,
                              const float* __restrict__ dmax, float* __restrict__ score,
                              int bbCount, int Nv, int ld) {
  int t = blockIdx.x * 256 + threadIdx.x;
  if (t >= bbCount * Nv) return;
  int bb = t / Nv, i = t % Nv;
  float di = dens[(long long)bb * ld + i];
  const float* row = dist + (long long)bb * ld * ld + (long long)i * ld;
  float best = dmax[bb];
  for (int j = 0; j < Nv; j++)
    if (dens[(long long)bb * ld + j] > di) best = fminf(best, row[j]);
  score[(long long)bb * ld + i] = best * di;
}
__global__ void topk_seeds(const float* __restrict__ score, int* __restrict__ seeds,
                           int bbCount, int Nv, int ld, int K) {
  int bb = blockIdx.x * 64 + threadIdx.x;
  if (bb >= bbCount) return;
  const float* s = score + (long long)bb * ld;
  unsigned char taken[224] = {0};
  for (int k = 0; k < K; k++) {
    float b = -3e38f; int bi = 0;
    for (int j = 0; j < Nv; j++)
      if (!taken[j] && s[j] > b) { b = s[j]; bi = j; }
    taken[bi] = 1;
    seeds[bb * K + k] = bi;
  }
}
__global__ void assign_clusters(const float* __restrict__ dist, const int* __restrict__ seeds,
                                int* __restrict__ idx, int bbCount, int Nv, int ld, int K) {
  int t = blockIdx.x * 256 + threadIdx.x;
  if (t >= bbCount * Nv) return;
  int bb = t / Nv, n = t % Nv;
  float b = 3e38f; int bk = 0;
  for (int k = 0; k < K; k++) {
    float d = dist[(long long)bb * ld * ld + (long long)seeds[bb * K + k] * ld + n];
    if (d < b) { b = d; bk = k; }
  }
  idx[(long long)bb * ld + n] = bk;
}
__global__ void seed_override(const int* __restrict__ seeds, int* __restrict__ idx,
                              int bbCount, int ld, int K) {
  int t = blockIdx.x * 256 + threadIdx.x;
  if (t >= bbCount * K) return;
  int bb = t / K, k = t % K;
  idx[(long long)bb * ld + seeds[bb * K + k]] = k;
}
__global__ void count_members(const int* __restrict__ idx, float* __restrict__ cnt,
                              int bbCount, int Nv, int ldIdx, int ldCnt) {
  int t = blockIdx.x * 256 + threadIdx.x;
  if (t >= bbCount * Nv) return;
  int bb = t / Nv, n = t % Nv;
  atomicAdd(&cnt[bb * ldCnt + idx[(long long)bb * ldIdx + n]], 1.f);
}
__global__ void merge_scatter_t(const float* __restrict__ x, const int* __restrict__ idxT,
                                const float* __restrict__ cnt, float* __restrict__ merged) {
  long long i = blockIdx.x * 256ll + threadIdx.x;
  const long long total = (long long)C_ * B_ * N_ * D_;
  if (i >= total) return;
  int d = (int)(i & 511); long long r = i >> 9;
  int n = (int)(r % N_); int bb = (int)(r / N_);
  int b = bb & 3, c = bb >> 2;
  int k = idxT[(long long)bb * NP_ + n];
  float w = cnt[bb * KT_ + k] + 1e-6f;
  float v = x[(((long long)b * L_ + 1 + n) * C_ + c) * D_ + d] / w;
  atomicAdd(&merged[((long long)bb * KT_ + k) * D_ + d], v);
}
__global__ void gather_xin(const float* __restrict__ x, const float* __restrict__ merged,
                           const int* __restrict__ idxT, _Float16* __restrict__ xinh) {
  long long i = blockIdx.x * 256ll + threadIdx.x;
  const long long total = (long long)B_ * L_ * C_ * D_;
  if (i >= total) return;
  int d = (int)(i & 511); long long r = i >> 9;
  int c = (int)(r % C_); r /= C_;
  int l = (int)(r % L_); int b = (int)(r / L_);
  float v;
  if (l == 0) v = x[(((long long)b * L_) * C_ + c) * D_ + d];
  else {
    int bb = c * B_ + b;
    int k = idxT[(long long)bb * NP_ + (l - 1)];
    v = merged[((long long)bb * KT_ + k) * D_ + d];
  }
  xinh[i] = (_Float16)v;
}
__global__ void merge_scatter_c(const float* __restrict__ x, const int* __restrict__ idxC,
                                const float* __restrict__ cnt, float* __restrict__ merged) {
  long long i = blockIdx.x * 256ll + threadIdx.x;
  const long long total = (long long)B_ * L_ * C_ * D_;
  if (i >= total) return;
  int d = (int)(i & 511); long long r = i >> 9;
  int c = (int)(r % C_); r /= C_;
  int l = (int)(r % L_); int b = (int)(r / L_);
  int k = idxC[b * CP_ + c];
  float w = cnt[b * 16 + k] + 1e-6f;
  atomicAdd(&merged[(((long long)b * L_ + l) * KC_ + k) * D_ + d], x[i] / w);
}
__global__ void gather_xc(const float* __restrict__ merged, const int* __restrict__ idxC,
                          _Float16* __restrict__ out) {
  long long i = blockIdx.x * 256ll + threadIdx.x;
  const long long total = (long long)B_ * C_ * L_ * D_;
  if (i >= total) return;
  int d = (int)(i & 511); long long r = i >> 9;
  int l = (int)(r % L_); r /= L_;
  int c = (int)(r % C_); int b = (int)(r / C_);
  int k = idxC[b * CP_ + c];
  out[i] = (_Float16)merged[(((long long)b * L_ + l) * KC_ + k) * D_ + d];
}

// ------------------------------------------------------------ attention bits
__global__ void transpose_pad(const _Float16* __restrict__ V, _Float16* __restrict__ Vt,
                              int batches, int rows, int rowsPad, int cols,
                              long long inOuter, long long inInner, int innerN,
                              long long inRowStride, long long outStride) {
  long long i = blockIdx.x * 256ll + threadIdx.x;
  long long total = (long long)batches * cols * rowsPad;
  if (i >= total) return;
  int lp = (int)(i % rowsPad); long long r = i / rowsPad;
  int dcol = (int)(r % cols); int bb = (int)(r / cols);
  long long ioff = (long long)(bb / innerN) * inOuter + (long long)(bb % innerN) * inInner;
  _Float16 v = (lp < rows) ? V[ioff + (long long)lp * inRowStride + dcol] : (_Float16)0.f;
  Vt[(long long)bb * outStride + (long long)dcol * rowsPad + lp] = v;
}
__global__ void softmax_k(const float* __restrict__ S, _Float16* __restrict__ P,
                          float* __restrict__ attnOut, int batches, int Mp, int ldS,
                          int ldP, int Mvalid, int Tvalid, float scale, int Lr,
                          int writeAttn) {
  int wid = (blockIdx.x * blockDim.x + threadIdx.x) >> 5;
  int lane = threadIdx.x & 31;
  if (wid >= batches * Mp) return;
  int bb = wid / Mp, i = wid % Mp;
  _Float16* prow = P + (long long)bb * Mp * ldP + (long long)i * ldP;
  if (i >= Mvalid) {                       // wave-uniform branch
    for (int j = lane; j < ldP; j += 32) prow[j] = (_Float16)0.f;
    return;
  }
  const float* srow = S + (long long)bb * Mp * ldS + (long long)i * ldS;
  float m = -3e38f;
  for (int j = lane; j < Tvalid; j += 32) m = fmaxf(m, srow[j]);
  for (int o = 16; o > 0; o >>= 1) m = fmaxf(m, __shfl_xor(m, o, 32));
  float s = 0.f;
  for (int j = lane; j < Tvalid; j += 32) s += __expf((srow[j] - m) * scale);
  for (int o = 16; o > 0; o >>= 1) s += __shfl_xor(s, o, 32);
  float inv = 1.f / s;
  for (int j = lane; j < ldP; j += 32) {
    float p = (j < Tvalid) ? __expf((srow[j] - m) * scale) * inv : 0.f;
    prow[j] = (_Float16)p;
    if (writeAttn && j < Tvalid)
      attnOut[((long long)bb * Lr + i) * Lr + j] = p;
  }
}

// ------------------------------------------------------------ LN / residuals
__global__ void combine_ln1(const float* __restrict__ x, const float* __restrict__ nx1,
                            const float* __restrict__ nx2, const float* __restrict__ g,
                            const float* __restrict__ be, float* __restrict__ xmid,
                            _Float16* __restrict__ xmidh) {
  int tok = blockIdx.x;                      // (b*L + l)*C + c
  int c = tok % C_; int bl = tok / C_;
  int l = bl % L_;  int b = bl / L_;
  long long xoff = (long long)tok * D_;
  long long o1 = ((long long)(b * C_ + c) * NP_ + l) * D_;
  long long o2 = ((long long)(b * L_ + l) * CP_ + c) * D_;
  int t = threadIdx.x;
  float v0 = x[xoff + t]       + nx1[o1 + t]       * nx2[o2 + t];
  float v1 = x[xoff + t + 256] + nx1[o1 + t + 256] * nx2[o2 + t + 256];
  __shared__ float ssum[256], ssq[256];
  ssum[t] = v0 + v1; ssq[t] = v0 * v0 + v1 * v1; __syncthreads();
  for (int s = 128; s > 0; s >>= 1) {
    if (t < s) { ssum[t] += ssum[t + s]; ssq[t] += ssq[t + s]; }
    __syncthreads();
  }
  float mean = ssum[0] / (float)D_;
  float var = ssq[0] / (float)D_ - mean * mean;
  float inv = rsqrtf(var + 1e-5f);
  float r0 = (v0 - mean) * inv * g[t] + be[t];
  float r1 = (v1 - mean) * inv * g[t + 256] + be[t + 256];
  xmid[xoff + t] = r0;        xmid[xoff + t + 256] = r1;
  xmidh[xoff + t] = (_Float16)r0; xmidh[xoff + t + 256] = (_Float16)r1;
}
__global__ void final_ln2(const float* __restrict__ xmid, const float* __restrict__ mlpo,
                          const float* __restrict__ g, const float* __restrict__ be,
                          float* __restrict__ y) {
  int tok = blockIdx.x;
  long long off = (long long)tok * D_;
  int t = threadIdx.x;
  float v0 = xmid[off + t]       + mlpo[off + t];
  float v1 = xmid[off + t + 256] + mlpo[off + t + 256];
  __shared__ float ssum[256], ssq[256];
  ssum[t] = v0 + v1; ssq[t] = v0 * v0 + v1 * v1; __syncthreads();
  for (int s = 128; s > 0; s >>= 1) {
    if (t < s) { ssum[t] += ssum[t + s]; ssq[t] += ssq[t + s]; }
    __syncthreads();
  }
  float mean = ssum[0] / (float)D_;
  float var = ssq[0] / (float)D_ - mean * mean;
  float inv = rsqrtf(var + 1e-5f);
  y[off + t]       = (v0 - mean) * inv * g[t] + be[t];
  y[off + t + 256] = (v1 - mean) * inv * g[t + 256] + be[t + 256];
}

// ================================================================= host side
static inline int nblk(long long n) { return (int)((n + 255) / 256); }

extern "C" void kernel_launch(void* const* d_in, const int* in_sizes, int n_in,
                              void* d_out, int out_size, void* d_ws, size_t ws_size,
                              hipStream_t stream) {
  (void)in_sizes; (void)n_in; (void)out_size;
  const float* x   = (const float*)d_in[0];
  const float* Wq1 = (const float*)d_in[1];
  const float* Wk1 = (const float*)d_in[2];
  const float* Wv1 = (const float*)d_in[3];
  const float* Wo1 = (const float*)d_in[4];
  const float* Wq2 = (const float*)d_in[5];
  const float* Wk2 = (const float*)d_in[6];
  const float* Wv2 = (const float*)d_in[7];
  const float* Wo2 = (const float*)d_in[8];
  const float* W3  = (const float*)d_in[9];
  const float* b3  = (const float*)d_in[10];
  const float* W4  = (const float*)d_in[11];
  const float* b4  = (const float*)d_in[12];
  const float* g1  = (const float*)d_in[13];
  const float* be1 = (const float*)d_in[14];
  const float* g2  = (const float*)d_in[15];
  const float* be2 = (const float*)d_in[16];
  float* yout = (float*)d_out;

  const long long Y_CNT = (long long)B_ * L_ * C_ * D_;      // 22,640,640
  const long long ATT_CNT = (long long)B_ * C_ * L_ * L_;    // 8,888,220
  const long long SIG_CNT = (long long)B_ * L_ * C_;         // 44,220
  float* attn_out = yout + Y_CNT;
  float* mask_out = attn_out + ATT_CNT;
  float* sig_out  = mask_out + ATT_CNT;

  // ---- workspace bump allocator
  size_t off = 0;
  auto alloc = [&](size_t bytes) -> void* {
    void* p = (char*)d_ws + off;
    off += (bytes + 255) & ~(size_t)255;
    return p;
  };
  _Float16* w8t[8];
  for (int i = 0; i < 8; i++) w8t[i] = (_Float16*)alloc((size_t)D_ * D_ * 2);
  _Float16* w3t = (_Float16*)alloc((size_t)F_ * D_ * 2);
  _Float16* w4t = (_Float16*)alloc((size_t)D_ * F_ * 2);
  _Float16* xch = (_Float16*)alloc((size_t)C_ * B_ * NP_ * D_ * 2);
  _Float16* xc0h = (_Float16*)alloc((size_t)B_ * CP_ * D_ * 2);
  _Float16* xinh = (_Float16*)alloc((size_t)ROWS_CAP * D_ * 2);   // xin / xc shared
  _Float16* qh = (_Float16*)alloc((size_t)ROWS_CAP * D_ * 2);
  _Float16* kh = (_Float16*)alloc((size_t)ROWS_CAP * D_ * 2);
  _Float16* vh = (_Float16*)alloc((size_t)ROWS_CAP * D_ * 2);
  _Float16* vth = (_Float16*)alloc((size_t)26345472 * 2);  // max(220*512*224, 804*512*64)
  _Float16* ph  = (_Float16*)alloc((size_t)10250240 * 2);  // max(220*208*224, 804*64*64)
  _Float16* oh  = (_Float16*)alloc((size_t)26345472 * 2);  // max(220*208*512, 804*64*512)
  _Float16* xmidh = (_Float16*)alloc((size_t)MFLAT * D_ * 2);
  _Float16* y1h = (_Float16*)alloc((size_t)MFLAT * F_ * 2);
  float* gbuf = (float*)alloc((size_t)220 * NP_ * NP_ * 4);   // Gram -> scores reuse
  float* dbuf = (float*)alloc((size_t)220 * NP_ * NP_ * 4);   // dist
  float* dens = (float*)alloc((size_t)220 * NP_ * 4);
  float* dmax = (float*)alloc((size_t)220 * 4);
  float* scoreT = (float*)alloc((size_t)220 * NP_ * 4);
  int*   seedsT = (int*)alloc((size_t)220 * KT_ * 4);
  int*   idxT   = (int*)alloc((size_t)220 * NP_ * 4);
  float* cntT   = (float*)alloc((size_t)220 * KT_ * 4);
  float* mergT  = (float*)alloc((size_t)220 * KT_ * D_ * 4);
  float* gc   = (float*)alloc((size_t)B_ * CP_ * CP_ * 4);
  float* dc   = (float*)alloc((size_t)B_ * CP_ * CP_ * 4);
  float* densC = (float*)alloc((size_t)B_ * CP_ * 4);
  float* dmaxC = (float*)alloc((size_t)B_ * 4);
  float* scoreC = (float*)alloc((size_t)B_ * CP_ * 4);
  int*   seedsC = (int*)alloc((size_t)B_ * 16 * 4);
  int*   idxC   = (int*)alloc((size_t)B_ * CP_ * 4);
  float* cntC   = (float*)alloc((size_t)B_ * 16 * 4);
  float* mergC  = (float*)alloc((size_t)B_ * L_ * KC_ * D_ * 4);
  float* nx1  = (float*)alloc((size_t)MT1 * D_ * 4);
  float* nx2  = (float*)alloc((size_t)MX2 * D_ * 4);
  float* xmid = (float*)alloc((size_t)B_ * L_ * C_ * D_ * 4);
  float* mlpo = (float*)alloc((size_t)MFLAT * D_ * 4);
  if (off > ws_size) return;   // not enough scratch; abstain

  auto gemmL = [&](const _Float16* A, const _Float16* Bt, void* C, const float* bias,
                   int M, int N, int K, int lda, int ldb, int ldc,
                   long long aO, long long aI, int aN,
                   long long bO, long long bI, int bN,
                   long long cS, int batches, int flags, float alpha) {
    int Mt = M / 16, Nt = N / 16;
    long long total = (long long)batches * Mt * Nt;
    int blocks = (int)((total + 7) / 8);
    gemm_nt<<<blocks, 256, 0, stream>>>(A, Bt, C, bias, Mt, Nt, K, lda, ldb, ldc,
                                        aO, aI, aN, bO, bI, bN, cS, batches, flags, alpha);
  };
  auto gemm2L = [&](const _Float16* A, const _Float16* Bt, void* C, const float* bias,
                    int M, int N, int K, int lda, int ldb, int ldc, int flags, float alpha) {
    int Mt2 = M / 32, Nt2 = N / 32;
    int blocks = (Mt2 * Nt2 + 7) / 8;
    gemm_nt2<<<blocks, 256, 0, stream>>>(A, Bt, C, bias, Mt2, Nt2, K, lda, ldb, ldc, flags, alpha);
  };

  // ---- 0: weights f32 -> f16 (transposed to [N,K]) via LDS (async DMA if available)
  const float* wsq[8] = {Wq1, Wk1, Wv1, Wo1, Wq2, Wk2, Wv2, Wo2};
  for (int i = 0; i < 8; i++)
    wt_transpose_lds<<<(D_ / 64) * (D_ / 64), 256, 0, stream>>>(wsq[i], w8t[i], D_, D_);
  wt_transpose_lds<<<(D_ / 64) * (F_ / 64), 256, 0, stream>>>(W3, w3t, D_, F_);
  wt_transpose_lds<<<(F_ / 64) * (D_ / 64), 256, 0, stream>>>(W4, w4t, F_, D_);
  // constant outputs
  fill_f32k<<<nblk(ATT_CNT), 256, 0, stream>>>(mask_out, ATT_CNT, 0.f);
  fill_f32k<<<nblk(SIG_CNT), 256, 0, stream>>>(sig_out, SIG_CNT, 0.f);

  // ---- 1: temporal DPC-KNN clustering (per channel)
  build_xch_t<<<nblk((long long)C_ * B_ * NP_ * D_), 256, 0, stream>>>(x, xch);
  gemmL(xch, xch, gbuf, nullptr, NP_, NP_, D_, D_, D_, NP_,
        (long long)NP_ * D_, 0, 1, (long long)NP_ * D_, 0, 1,
        (long long)NP_ * NP_, 220, 0, 1.f);                                  // Gram via WMMA
  dist_from_gram<<<nblk((long long)220 * NP_ * NP_), 256, 0, stream>>>(gbuf, dbuf, 220, N_, NP_);
  knn_density<<<nblk(220 * N_), 256, 0, stream>>>(dbuf, dens, 220, N_, NP_, 20, 0x9E3779B9u);
  reduce_maxk<<<220, 256, 0, stream>>>(dbuf, dmax, N_, NP_);
  dparent_score<<<nblk(220 * N_), 256, 0, stream>>>(dbuf, dens, dmax, scoreT, 220, N_, NP_);
  topk_seeds<<<nblk(220), 64, 0, stream>>>(scoreT, seedsT, 220, N_, NP_, KT_);
  assign_clusters<<<nblk(220 * N_), 256, 0, stream>>>(dbuf, seedsT, idxT, 220, N_, NP_, KT_);
  seed_override<<<nblk(220 * KT_), 256, 0, stream>>>(seedsT, idxT, 220, NP_, KT_);
  fill_f32k<<<nblk(220 * KT_), 256, 0, stream>>>(cntT, 220 * KT_, 0.f);
  count_members<<<nblk(220 * N_), 256, 0, stream>>>(idxT, cntT, 220, N_, NP_, KT_);
  fill_f32k<<<nblk((long long)220 * KT_ * D_), 256, 0, stream>>>(mergT, (long long)220 * KT_ * D_, 0.f);
  merge_scatter_t<<<nblk((long long)220 * N_ * D_), 256, 0, stream>>>(x, idxT, cntT, mergT);
  fill_f16k<<<nblk((long long)ROWS_CAP * D_), 256, 0, stream>>>(xinh, (long long)ROWS_CAP * D_);
  gather_xin<<<nblk(Y_CNT), 256, 0, stream>>>(x, mergT, idxT, xinh);

  // ---- 2: temporal attention (batches bb = b*C + c, seq = L)
  long long tailN = (long long)(ROWS_CAP - MFLAT) * D_;
  fill_f16k<<<nblk(tailN), 256, 0, stream>>>(qh + (long long)MFLAT * D_, tailN);
  fill_f16k<<<nblk(tailN), 256, 0, stream>>>(kh + (long long)MFLAT * D_, tailN);
  fill_f16k<<<nblk(tailN), 256, 0, stream>>>(vh + (long long)MFLAT * D_, tailN);
  gemm2L(xinh, w8t[0], qh, nullptr, MFLAT, D_, D_, D_, D_, D_, EP_F16, 1.f);
  gemm2L(xinh, w8t[1], kh, nullptr, MFLAT, D_, D_, D_, D_, D_, EP_F16, 1.f);
  gemm2L(xinh, w8t[2], vh, nullptr, MFLAT, D_, D_, D_, D_, D_, EP_F16, 1.f);
  transpose_pad<<<nblk((long long)220 * D_ * TPAD_), 256, 0, stream>>>(
      vh, vth, 220, L_, TPAD_, D_,
      (long long)L_ * C_ * D_, D_, C_, (long long)C_ * D_, (long long)D_ * TPAD_);
  gemmL(qh, kh, gbuf, nullptr, NP_, NP_, D_, C_ * D_, C_ * D_, NP_,
        (long long)L_ * C_ * D_, D_, C_, (long long)L_ * C_ * D_, D_, C_,
        (long long)NP_ * NP_, 220, 0, 1.f);                                  // scores
  softmax_k<<<nblk((long long)220 * NP_ * 32), 256, 0, stream>>>(
      gbuf, ph, attn_out, 220, NP_, NP_, TPAD_, L_, L_, INV_SQRT_D, L_, 1);
  gemmL(ph, vth, oh, nullptr, NP_, D_, TPAD_, TPAD_, TPAD_, D_,
        (long long)NP_ * TPAD_, 0, 1, (long long)D_ * TPAD_, 0, 1,
        (long long)NP_ * D_, 220, EP_F16, 1.f);                              // P @ V
  gemm2L(oh, w8t[3], nx1, nullptr, MT1, D_, D_, D_, D_, D_, 0, 1.f);

  // ---- 3: channel clustering on x[:,0]
  build_xc0<<<nblk(B_ * CP_ * D_), 256, 0, stream>>>(x, xc0h);
  gemmL(xc0h, xc0h, gc, nullptr, CP_, CP_, D_, D_, D_, CP_,
        (long long)CP_ * D_, 0, 1, (long long)CP_ * D_, 0, 1,
        (long long)CP_ * CP_, B_, 0, 1.f);
  dist_from_gram<<<nblk((long long)B_ * CP_ * CP_), 256, 0, stream>>>(gc, dc, B_, C_, CP_);
  knn_density<<<nblk(B_ * C_), 256, 0, stream>>>(dc, densC, B_, C_, CP_, 5, 0x85EBCA6Bu);
  reduce_maxk<<<B_, 256, 0, stream>>>(dc, dmaxC, C_, CP_);
  dparent_score<<<nblk(B_ * C_), 256, 0, stream>>>(dc, densC, dmaxC, scoreC, B_, C_, CP_);
  topk_seeds<<<1, 64, 0, stream>>>(scoreC, seedsC, B_, C_, CP_, KC_);
  assign_clusters<<<nblk(B_ * C_), 256, 0, stream>>>(dc, seedsC, idxC, B_, C_, CP_, KC_);
  seed_override<<<nblk(B_ * KC_), 256, 0, stream>>>(seedsC, idxC, B_, CP_, KC_);
  fill_f32k<<<nblk(B_ * 16), 256, 0, stream>>>(cntC, B_ * 16, 0.f);
  count_members<<<nblk(B_ * C_), 256, 0, stream>>>(idxC, cntC, B_, C_, CP_, 16);
  fill_f32k<<<nblk((long long)B_ * L_ * KC_ * D_), 256, 0, stream>>>(mergC, (long long)B_ * L_ * KC_ * D_, 0.f);
  merge_scatter_c<<<nblk(Y_CNT), 256, 0, stream>>>(x, idxC, cntC, mergC);
  fill_f16k<<<nblk((long long)ROWS_CAP * D_), 256, 0, stream>>>(xinh, (long long)ROWS_CAP * D_);
  gather_xc<<<nblk(Y_CNT), 256, 0, stream>>>(mergC, idxC, xinh);

  // ---- 4: channel attention (batches bb = b*L + l, seq = C)
  gemm2L(xinh, w8t[4], qh, nullptr, MFLAT, D_, D_, D_, D_, D_, EP_F16, 1.f);
  gemm2L(xinh, w8t[5], kh, nullptr, MFLAT, D_, D_, D_, D_, D_, EP_F16, 1.f);
  gemm2L(xinh, w8t[6], vh, nullptr, MFLAT, D_, D_, D_, D_, D_, EP_F16, 1.f);
  transpose_pad<<<nblk((long long)804 * D_ * CP_), 256, 0, stream>>>(
      vh, vth, 804, C_, CP_, D_,
      (long long)C_ * L_ * D_, D_, L_, (long long)L_ * D_, (long long)D_ * CP_);
  gemmL(qh, kh, gbuf, nullptr, CP_, CP_, D_, L_ * D_, L_ * D_, CP_,
        (long long)C_ * L_ * D_, D_, L_, (long long)C_ * L_ * D_, D_, L_,
        (long long)CP_ * CP_, 804, 0, 1.f);
  softmax_k<<<nblk((long long)804 * CP_ * 32), 256, 0, stream>>>(
      gbuf, ph, nullptr, 804, CP_, CP_, CP_, C_, C_, INV_SQRT_D, C_, 0);
  gemmL(ph, vth, oh, nullptr, CP_, D_, CP_, CP_, CP_, D_,
        (long long)CP_ * CP_, 0, 1, (long long)D_ * CP_, 0, 1,
        (long long)CP_ * D_, 804, EP_F16, 1.f);
  gemm2L(oh, w8t[7], nx2, nullptr, MX2, D_, D_, D_, D_, D_, 0, 1.f);

  // ---- 5: residual combine + LN1 + MLP + LN2
  combine_ln1<<<(int)SIG_CNT, 256, 0, stream>>>(x, nx1, nx2, g1, be1, xmid, xmidh);
  fill_f16k<<<nblk((long long)(MFLAT - SIG_CNT) * D_), 256, 0, stream>>>(
      xmidh + SIG_CNT * D_, (long long)(MFLAT - SIG_CNT) * D_);
  gemm2L(xmidh, w3t, y1h, b3, MFLAT, F_, D_, D_, D_, F_, EP_F16 | EP_BIAS | EP_GELU, 1.f);
  gemm2L(y1h, w4t, mlpo, b4, MFLAT, D_, F_, F_, F_, D_, EP_BIAS, 1.f);
  final_ln2<<<(int)SIG_CNT, 256, 0, stream>>>(xmid, mlpo, g2, be2, yout);
}
